// GeneralCausalSelfAttention_62929860821355
// MI455X (gfx1250) — matrix operand order repaired and checked by reading, not compiled, not run on hardware
//
#include <hip/hip_runtime.h>
#include <stdint.h>

// ---------------- problem constants ----------------
constexpr int Bb   = 2;
constexpr int Tn   = 1024;     // tokens == concepts
constexpr int NEc  = 1024;
constexpr int CDc  = 1024;
constexpr int NHc  = 16;
constexpr int HS   = 64;       // head size (== hsc)
constexpr int OUTD = NEc + 3 * CDc;  // 4096

typedef __attribute__((ext_vector_type(16))) __bf16 bf16x16;
typedef __attribute__((ext_vector_type(8)))  float  f32x8;

union FragAB {
  uint4   u4[2];
  bf16x16 v;
};

static __device__ __forceinline__ unsigned short f2bf(float x) {
  unsigned u = __float_as_uint(x);
  u += 0x7FFFu + ((u >> 16) & 1u);           // round-to-nearest-even
  return (unsigned short)(u >> 16);
}

static __device__ __forceinline__ void wait_ds0() {
#if __has_builtin(__builtin_amdgcn_s_wait_dscnt)
  __builtin_amdgcn_s_wait_dscnt(0);
#else
  asm volatile("s_wait_dscnt 0" ::: "memory");
#endif
}

template <int N>
static __device__ __forceinline__ void wait_async() {
#if __has_builtin(__builtin_amdgcn_s_wait_asynccnt)
  __builtin_amdgcn_s_wait_asynccnt(N);
#else
  asm volatile("s_wait_asynccnt %0" ::: "memory");
#endif
}
#if !__has_builtin(__builtin_amdgcn_s_wait_asynccnt)
template <>
__device__ __forceinline__ void wait_async<0>() { asm volatile("s_wait_asynccnt 0" ::: "memory"); }
template <>
__device__ __forceinline__ void wait_async<4>() { asm volatile("s_wait_asynccnt 4" ::: "memory"); }
template <>
__device__ __forceinline__ void wait_async<8>() { asm volatile("s_wait_asynccnt 8" ::: "memory"); }
#endif

// async Global -> LDS, 16 bytes per lane, tracked by ASYNCcnt
static __device__ __forceinline__ void async_g2l_b128(unsigned lds_off,
                                                      const void* gaddr) {
  asm volatile("global_load_async_to_lds_b128 %0, %1, off"
               :: "v"(lds_off), "v"(gaddr)
               : "memory");
}

// ---------------- fp32 -> bf16 convert ----------------
__global__ void k_f32_to_bf16(const float* __restrict__ in,
                              unsigned short* __restrict__ out, int n) {
  int i = blockIdx.x * blockDim.x + threadIdx.x;
  int stride = gridDim.x * blockDim.x;
  for (; i < n; i += stride) out[i] = f2bf(in[i]);
}

// ---------------- WMMA GEMM: C(M,N) = A(M,K) * W(N,K)^T + bias ----------------
// block = 256 threads (8 waves), tile 128x128, wave tile 32x64.
// Double-buffered async Global->LDS staging of 128x32 A and B tiles.
template <bool OUT_BF16>
__global__ __launch_bounds__(256) void k_gemm_wmma(
    const unsigned short* __restrict__ A,   // (M,K) bf16 row-major
    const unsigned short* __restrict__ W,   // (N,K) bf16 row-major
    const float* __restrict__ bias,         // (N)
    void* __restrict__ Cout, int M, int N, int K) {
  // [buf][0=A,1=B][row * 48 + k]   row stride 48 elems (96B: 16B-aligned rows)
  __shared__ unsigned short smem[2][2][128 * 48];

  const int tid  = threadIdx.x;
  const int lane = tid & 31;
  const int wave = tid >> 5;
  const int wm = wave >> 1;               // 0..3
  const int wn = wave & 1;                // 0..1
  const int rBase = blockIdx.y * 128;
  const int cBase = blockIdx.x * 128;
  const int lm = lane & 15;
  const int lh = lane >> 4;

  const unsigned smem_base = (unsigned)(uintptr_t)(&smem[0][0][0]);

  // cooperative async tile fetch: 128 rows x 32 k (bf16) = 512 x 16B chunks
  auto load_tile = [&](const unsigned short* src, int row0, int k0, int buf, int ab) {
#pragma unroll
    for (int j = 0; j < 2; j++) {
      const int c   = tid * 2 + j;        // 0..511
      const int row = c >> 2;
      const int kk  = (c & 3) * 8;
      const unsigned short* g = src + (size_t)(row0 + row) * K + k0 + kk;
      const unsigned loff =
          smem_base + (unsigned)(((buf * 2 + ab) * (128 * 48) + row * 48 + kk) * 2);
      async_g2l_b128(loff, (const void*)g);
    }
  };

  f32x8 acc[2][4];
#pragma unroll
  for (int i = 0; i < 2; i++)
#pragma unroll
    for (int j = 0; j < 4; j++)
#pragma unroll
      for (int v = 0; v < 8; v++) acc[i][j][v] = 0.0f;

  load_tile(A, rBase, 0, 0, 0);
  load_tile(W, cBase, 0, 0, 1);

  const int nk = K / 32;
  for (int kt = 0; kt < nk; kt++) {
    const int buf = kt & 1;
    if (kt + 1 < nk) {
      load_tile(A, rBase, (kt + 1) * 32, buf ^ 1, 0);
      load_tile(W, cBase, (kt + 1) * 32, buf ^ 1, 1);
      wait_async<4>();   // 4 in-flight (next tile); current tile complete (in-order)
    } else {
      wait_async<0>();
    }
    __syncthreads();     // whole tile visible to all waves

    const unsigned short* As = smem[buf][0];
    const unsigned short* Bs = smem[buf][1];
    FragAB a[2], b[4];
#pragma unroll
    for (int i = 0; i < 2; i++) {
      // 16-bit A 16x32 layout: lane holds K {8lh..8lh+7, 16+8lh..16+8lh+7}
      const unsigned short* p = As + (wm * 32 + 16 * i + lm) * 48 + 8 * lh;
      a[i].u4[0] = *(const uint4*)(p);
      a[i].u4[1] = *(const uint4*)(p + 16);
    }
#pragma unroll
    for (int j = 0; j < 4; j++) {
      // 16-bit B 32x16 layout: lane n holds K {8lh..8lh+7, 16+8lh..16+8lh+7}
      const unsigned short* p = Bs + (wn * 64 + 16 * j + lm) * 48 + 8 * lh;
      b[j].u4[0] = *(const uint4*)(p);
      b[j].u4[1] = *(const uint4*)(p + 16);
    }
#pragma unroll
    for (int i = 0; i < 2; i++)
#pragma unroll
      for (int j = 0; j < 4; j++)
        acc[i][j] = __builtin_amdgcn_wmma_f32_16x16x32_bf16(
            false, a[i].v, false, b[j].v, (short)0, acc[i][j], false, false);

    __syncthreads();     // all reads done before next iteration overwrites buf^1
  }

  const int rW = rBase + wm * 32;
  const int cW = cBase + wn * 64;
#pragma unroll
  for (int j = 0; j < 4; j++) {
    const int c = cW + 16 * j + lm;
    const float bv = bias[c];
#pragma unroll
    for (int i = 0; i < 2; i++) {
      const int rb = rW + 16 * i + 8 * lh;   // f32 D layout: row = vgpr (+8 hi half)
#pragma unroll
      for (int vg = 0; vg < 8; vg++) {
        const float v = acc[i][j][vg] + bv;
        const size_t off = (size_t)(rb + vg) * N + c;
        if (OUT_BF16)
          ((unsigned short*)Cout)[off] = f2bf(v);
        else
          ((float*)Cout)[off] = v;
      }
    }
  }
}

// ---------------- pack transposed V buffers: (B,NH,HS,2T) ----------------
__global__ void k_pack_vt(const unsigned short* __restrict__ tokb,
                          const unsigned short* __restrict__ conb,
                          unsigned short* __restrict__ vtokT,
                          unsigned short* __restrict__ vconT) {
  const int total = 1 << 23;  // 2 buffers * 2*16*64*2048
  for (int idx = blockIdx.x * blockDim.x + threadIdx.x; idx < total;
       idx += gridDim.x * blockDim.x) {
    const int buf = idx >> 22;
    const int rem = idx & ((1 << 22) - 1);
    const int s = rem & 2047;
    const int d = (rem >> 11) & 63;
    const int h = (rem >> 17) & 15;
    const int b = (rem >> 21) & 1;
    const int srow = s & (Tn - 1);
    const bool first = s < Tn;
    if (buf == 0) {  // v_tok = [vt_t(from tok) ; vc_t(from con)]
      const int col = NEc + CDc + h * HS + d;
      const unsigned short* src = first ? tokb : conb;
      vtokT[rem] = src[(size_t)(b * Tn + srow) * OUTD + col];
    } else {          // v_con = [vc_c(from con) ; vt_c(from tok)]
      const int col = NEc + 2 * CDc + h * HS + d;
      const unsigned short* src = first ? conb : tokb;
      vconT[rem] = src[(size_t)(b * Tn + srow) * OUTD + col];
    }
  }
}

// ---------------- flash attention (both sides), block-cooperative ----------------
// grid: (T/64, NH, B*2); block 128 (4 waves); wave = 16 query rows; block = 64 rows.
// Per 64-key chunk: async-stage K(64x64) and V^T(64x64) tiles to LDS (double
// buffered), then 16 WMMAs per wave reading LDS fragments.
__global__ __launch_bounds__(128) void k_attn(
    const unsigned short* __restrict__ xt_bf, const unsigned short* __restrict__ xc_bf,
    const unsigned short* __restrict__ tokb,  const unsigned short* __restrict__ conb,
    const unsigned short* __restrict__ vtokT, const unsigned short* __restrict__ vconT,
    unsigned short* __restrict__ yt_bf, unsigned short* __restrict__ yc_bf) {
  // K/V tiles: rows padded to 72 elems (144B) -> conflict-free, 16B-aligned
  __shared__ unsigned short kvs[2][2][64 * 72];
  __shared__ unsigned short pls[4][16][80];  // per-wave P bounce (stride 160B)

  const int tid  = threadIdx.x;
  const int lane = tid & 31;
  const int wave = tid >> 5;
  const int lm = lane & 15;
  const int lh = lane >> 4;
  const int q0 = blockIdx.x * 64;          // block query base
  const int qw0 = q0 + wave * 16;          // wave query base
  const int h = blockIdx.y;
  const int b = blockIdx.z >> 1;
  const int side = blockIdx.z & 1;         // 0 = token queries, 1 = concept queries

  const unsigned short* Q  = side ? xc_bf : xt_bf;
  const unsigned short* X1 = side ? conb : tokb;   // same-kind keys (causal)
  const unsigned short* X2 = side ? tokb : conb;   // cross-kind keys
  const unsigned short* VT = side ? vconT : vtokT;
  unsigned short* Y        = side ? yc_bf : yt_bf;
  const int colOff = (side ? NEc : 0) + h * HS;    // key column base in tok/con

  const unsigned kv_base = (unsigned)(uintptr_t)(&kvs[0][0][0]);

  // chunk schedule (uniform across the whole block)
  const int nA = q0 / 64 + 1;                               // same-kind chunks
  int limB;
  if (side == 0)
    limB = ((q0 + 63) >> 3) + 1;                            // concepts needed
  else {
    limB = 8 * q0 + 512;                                    // tokens needed
    if (limB > Tn) limB = Tn;
  }
  const int nB = (limB + 63) / 64;
  const int nCh = nA + nB;

  auto chunk_s = [&](int i) { return (i < nA) ? i * 64 : Tn + (i - nA) * 64; };

  // cooperative async staging of one chunk's K and V tiles
  auto stage = [&](int i, int buf) {
    const int sBase = chunk_s(i);
    const unsigned short* Xk = (i < nA) ? X1 : X2;
    const unsigned kbase = kv_base + (unsigned)((buf * 2 + 0) * (64 * 72) * 2);
    const unsigned vbase = kv_base + (unsigned)((buf * 2 + 1) * (64 * 72) * 2);
#pragma unroll
    for (int j = 0; j < 4; j++) {                 // K tile: 512 x 16B chunks
      const int c = tid * 4 + j;
      const int row = c >> 3;
      const int kk = (c & 7) * 8;
      const int srow = (sBase + row) & (Tn - 1);
      const unsigned short* g = Xk + (size_t)(b * Tn + srow) * OUTD + colOff + kk;
      async_g2l_b128(kbase + (unsigned)((row * 72 + kk) * 2), (const void*)g);
    }
#pragma unroll
    for (int j = 0; j < 4; j++) {                 // V^T tile: dims x keys
      const int c = tid * 4 + j;
      const int row = c >> 3;                     // dim 0..63
      const int kk = (c & 7) * 8;
      const unsigned short* g =
          VT + (size_t)((b * NHc + h) * HS + row) * (2 * Tn) + sBase + kk;
      async_g2l_b128(vbase + (unsigned)((row * 72 + kk) * 2), (const void*)g);
    }
  };

  // Q fragments (16x64 -> two 16x32 A-frags), loaded once
  FragAB aq[2];
  {
    const unsigned short* qp = Q + (size_t)(b * Tn + qw0 + lm) * NEc + h * HS + 8 * lh;
    aq[0].u4[0] = *(const uint4*)(qp);
    aq[0].u4[1] = *(const uint4*)(qp + 16);
    aq[1].u4[0] = *(const uint4*)(qp + 32);
    aq[1].u4[1] = *(const uint4*)(qp + 48);
  }

  f32x8 acc[4];
  float mrow[8], lrow[8];
#pragma unroll
  for (int j = 0; j < 4; j++)
#pragma unroll
    for (int v = 0; v < 8; v++) acc[j][v] = 0.0f;
#pragma unroll
  for (int v = 0; v < 8; v++) { mrow[v] = -3.0e38f; lrow[v] = 0.0f; }

  stage(0, 0);

  for (int i = 0; i < nCh; i++) {
    const int buf = i & 1;
    if (i + 1 < nCh) {
      stage(i + 1, buf ^ 1);
      wait_async<8>();   // 8 in-flight (next chunk); current chunk complete
    } else {
      wait_async<0>();
    }
    __syncthreads();

    const int sBase = chunk_s(i);
    const bool phaseB = (i >= nA);
    const unsigned short* Ks = kvs[buf][0];
    const unsigned short* Vs = kvs[buf][1];

    // ---- load all 8 K-fragments (independent regs -> pipelined ds loads)
    FragAB kf[4][2];
#pragma unroll
    for (int sub = 0; sub < 4; sub++) {
      const unsigned short* kp = Ks + (16 * sub + lm) * 72 + 8 * lh;
      kf[sub][0].u4[0] = *(const uint4*)(kp);
      kf[sub][0].u4[1] = *(const uint4*)(kp + 16);
      kf[sub][1].u4[0] = *(const uint4*)(kp + 32);
      kf[sub][1].u4[1] = *(const uint4*)(kp + 48);
    }
    // ---- scores: four 16x16 tiles (64 keys), K-dim 64 via two chained WMMAs
    f32x8 st[4];
#pragma unroll
    for (int sub = 0; sub < 4; sub++) {
      f32x8 z;
#pragma unroll
      for (int v = 0; v < 8; v++) z[v] = 0.0f;
      z = __builtin_amdgcn_wmma_f32_16x16x32_bf16(false, aq[0].v, false, kf[sub][0].v,
                                                  (short)0, z, false, false);
      z = __builtin_amdgcn_wmma_f32_16x16x32_bf16(false, aq[1].v, false, kf[sub][1].v,
                                                  (short)0, z, false, false);
      st[sub] = z;
    }
    // ---- scale + mask
#pragma unroll
    for (int sub = 0; sub < 4; sub++) {
      const int s = sBase + 16 * sub + lm;  // global key index [0,2048)
#pragma unroll
      for (int vg = 0; vg < 8; vg++) {
        const int q = qw0 + vg + 8 * lh;
        bool ok;
        if (!phaseB) {
          ok = (s <= q);                     // causal same-kind
        } else {
          const int t = s - Tn;
          ok = side ? ((t >> 3) <= q)        // concept->token: t < 8(q+1)
                    : (t <= (q >> 3));       // token->concept: c <= q/8
        }
        st[sub][vg] = ok ? st[sub][vg] * 0.125f : -1.0e9f;
      }
    }
    // ---- online softmax (row reductions across 16-lane groups)
    float corr[8];
#pragma unroll
    for (int vg = 0; vg < 8; vg++) {
      float mx = fmaxf(fmaxf(st[0][vg], st[1][vg]), fmaxf(st[2][vg], st[3][vg]));
#pragma unroll
      for (int off = 8; off > 0; off >>= 1) mx = fmaxf(mx, __shfl_xor(mx, off, 16));
      const float mnew = fmaxf(mrow[vg], mx);
      const float c = __expf(mrow[vg] - mnew);
      float e0 = __expf(st[0][vg] - mnew);
      float e1 = __expf(st[1][vg] - mnew);
      float e2 = __expf(st[2][vg] - mnew);
      float e3 = __expf(st[3][vg] - mnew);
      st[0][vg] = e0; st[1][vg] = e1; st[2][vg] = e2; st[3][vg] = e3;
      float ls = (e0 + e1) + (e2 + e3);
#pragma unroll
      for (int off = 8; off > 0; off >>= 1) ls += __shfl_xor(ls, off, 16);
      lrow[vg] = lrow[vg] * c + ls;
      mrow[vg] = mnew;
      corr[vg] = c;
    }
#pragma unroll
    for (int j = 0; j < 4; j++)
#pragma unroll
      for (int vg = 0; vg < 8; vg++) acc[j][vg] *= corr[vg];

    // ---- relayout P (D-layout) -> A-fragments via per-wave LDS bounce
#pragma unroll
    for (int sub = 0; sub < 4; sub++)
#pragma unroll
      for (int vg = 0; vg < 8; vg++)
        pls[wave][vg + 8 * lh][16 * sub + lm] = f2bf(st[sub][vg]);
    wait_ds0();
    FragAB ap[2];
#pragma unroll
    for (int kk = 0; kk < 2; kk++) {
      const unsigned short* lp = &pls[wave][lm][32 * kk + 8 * lh];
      ap[kk].u4[0] = *(const uint4*)(lp);
      ap[kk].u4[1] = *(const uint4*)(lp + 16);
    }
    // ---- load all 8 V-fragments, then O += P * V
    FragAB vf[4][2];
#pragma unroll
    for (int j = 0; j < 4; j++) {
      const unsigned short* vp = Vs + (16 * j + lm) * 72 + 8 * lh;
      vf[j][0].u4[0] = *(const uint4*)(vp);
      vf[j][0].u4[1] = *(const uint4*)(vp + 16);
      vf[j][1].u4[0] = *(const uint4*)(vp + 32);
      vf[j][1].u4[1] = *(const uint4*)(vp + 48);
    }
#pragma unroll
    for (int j = 0; j < 4; j++) {
      acc[j] = __builtin_amdgcn_wmma_f32_16x16x32_bf16(false, ap[0].v, false, vf[j][0].v,
                                                       (short)0, acc[j], false, false);
      acc[j] = __builtin_amdgcn_wmma_f32_16x16x32_bf16(false, ap[1].v, false, vf[j][1].v,
                                                       (short)0, acc[j], false, false);
    }

    __syncthreads();   // all waves done reading before buf^1 is overwritten
  }

  // ---- normalize + store (bf16, feeds output projection GEMM)
#pragma unroll
  for (int j = 0; j < 4; j++)
#pragma unroll
    for (int vg = 0; vg < 8; vg++) {
      const int q = qw0 + vg + 8 * lh;
      const float v = acc[j][vg] / lrow[vg];
      Y[(size_t)(b * Tn + q) * NEc + h * HS + 16 * j + lm] = f2bf(v);
    }
}

// ---------------- launch ----------------
extern "C" void kernel_launch(void* const* d_in, const int* in_sizes, int n_in,
                              void* d_out, int out_size, void* d_ws, size_t ws_size,
                              hipStream_t stream) {
  (void)in_sizes; (void)n_in; (void)out_size; (void)ws_size;
  const float* xt  = (const float*)d_in[0];
  const float* xc  = (const float*)d_in[1];
  const float* Wt  = (const float*)d_in[2];
  const float* bt  = (const float*)d_in[3];
  const float* Wc  = (const float*)d_in[4];
  const float* bc  = (const float*)d_in[5];
  const float* Wtp = (const float*)d_in[6];
  const float* btp = (const float*)d_in[7];
  const float* Wcp = (const float*)d_in[8];
  const float* bcp = (const float*)d_in[9];
  float* out = (float*)d_out;

  char* ws = (char*)d_ws;
  size_t off = 0;
  auto alloc = [&](size_t bytes) {
    void* p = ws + off;
    off += (bytes + 255) & ~(size_t)255;
    return p;
  };
  const size_t nXT  = (size_t)Bb * Tn * NEc;           // 2M
  const size_t nW   = (size_t)OUTD * NEc;              // 4M
  const size_t nWp  = (size_t)NEc * NEc;               // 1M
  const size_t nTok = (size_t)Bb * Tn * OUTD;          // 8M
  const size_t nVT  = (size_t)Bb * NHc * HS * 2 * Tn;  // 4M

  unsigned short* xt_bf  = (unsigned short*)alloc(nXT * 2);
  unsigned short* xc_bf  = (unsigned short*)alloc(nXT * 2);
  unsigned short* Wt_bf  = (unsigned short*)alloc(nW * 2);
  unsigned short* Wc_bf  = (unsigned short*)alloc(nW * 2);
  unsigned short* Wtp_bf = (unsigned short*)alloc(nWp * 2);
  unsigned short* Wcp_bf = (unsigned short*)alloc(nWp * 2);
  unsigned short* tok_bf = (unsigned short*)alloc(nTok * 2);
  unsigned short* con_bf = (unsigned short*)alloc(nTok * 2);
  unsigned short* vtokT  = (unsigned short*)alloc(nVT * 2);
  unsigned short* vconT  = (unsigned short*)alloc(nVT * 2);
  unsigned short* yt_bf  = (unsigned short*)alloc(nXT * 2);
  unsigned short* yc_bf  = (unsigned short*)alloc(nXT * 2);

  // 1) fp32 -> bf16
  k_f32_to_bf16<<<2048, 256, 0, stream>>>(xt,  xt_bf,  (int)nXT);
  k_f32_to_bf16<<<2048, 256, 0, stream>>>(xc,  xc_bf,  (int)nXT);
  k_f32_to_bf16<<<4096, 256, 0, stream>>>(Wt,  Wt_bf,  (int)nW);
  k_f32_to_bf16<<<4096, 256, 0, stream>>>(Wc,  Wc_bf,  (int)nW);
  k_f32_to_bf16<<<1024, 256, 0, stream>>>(Wtp, Wtp_bf, (int)nWp);
  k_f32_to_bf16<<<1024, 256, 0, stream>>>(Wcp, Wcp_bf, (int)nWp);

  // 2) input projections: tok = xt @ Wt^T + bt ; con = xc @ Wc^T + bc
  k_gemm_wmma<true><<<dim3(OUTD / 128, (Bb * Tn) / 128), 256, 0, stream>>>(
      xt_bf, Wt_bf, bt, tok_bf, Bb * Tn, OUTD, NEc);
  k_gemm_wmma<true><<<dim3(OUTD / 128, (Bb * Tn) / 128), 256, 0, stream>>>(
      xc_bf, Wc_bf, bc, con_bf, Bb * Tn, OUTD, CDc);

  // 3) transpose-pack V for both attentions
  k_pack_vt<<<4096, 256, 0, stream>>>(tok_bf, con_bf, vtokT, vconT);

  // 4) both attentions (token side + concept side)
  k_attn<<<dim3(Tn / 64, NHc, Bb * 2), 128, 0, stream>>>(
      xt_bf, xc_bf, tok_bf, con_bf, vtokT, vconT, yt_bf, yc_bf);

  // 5) output projections straight into d_out (y_t, then y_c)
  k_gemm_wmma<false><<<dim3(NEc / 128, (Bb * Tn) / 128), 256, 0, stream>>>(
      yt_bf, Wtp_bf, btp, out, Bb * Tn, NEc, NEc);
  k_gemm_wmma<false><<<dim3(CDc / 128, (Bb * Tn) / 128), 256, 0, stream>>>(
      yc_bf, Wcp_bf, bcp, out + (size_t)Bb * Tn * NEc, Bb * Tn, CDc, CDc);
}
